// ADNNx_25786983645309
// MI455X (gfx1250) — matrix operand order repaired and checked by reading, not compiled, not run on hardware
//
#include <hip/hip_runtime.h>
#include <hip/hip_bf16.h>

typedef __attribute__((ext_vector_type(16))) _Float16 v16h;
typedef __attribute__((ext_vector_type(8)))  float    v8f;

constexpr int kB = 32, kN = 256, kF = 128, kK = 32, kKNB = 16, kM = 4, kC = 8;
constexpr int kH = 256, kHP = 128, kHA = 64, kHE = 128;
constexpr int kPair = 257, kPairP = 288;          // K padded to multiple of 32
constexpr float kRC = 6.0f, kUPD = 0.5f;
constexpr float kINF = 3.0e38f;

static __device__ __forceinline__ float siluf(float x) {
  return x / (1.0f + __expf(-x));
}

// A fragment: 16x32 f16 from row-major storage (rows contiguous) -> 2x b128/lane
static __device__ __forceinline__ v16h loadA(const _Float16* base, int ld) {
  int lane = threadIdx.x & 31;
  int row  = lane & 15;
  int kb   = (lane & 16) ? 16 : 0;
  const _Float16* p = base + row * ld;
  v16h a;
#pragma unroll
  for (int v = 0; v < 8; ++v) {
    int k = kb + (v << 1);
    a[2 * v]     = p[k];
    a[2 * v + 1] = p[k + 1];
  }
  return a;
}

// B fragment: 32x16 f16 from K-major (transposed) storage BT[N][K] -> 16
// contiguous halves per lane (one 16B-aligned run of 32B -> b128 loads)
static __device__ __forceinline__ v16h loadBT(const _Float16* baseT, int ldT) {
  int lane = threadIdx.x & 31;
  int col  = lane & 15;
  int kb   = (lane & 16) ? 16 : 0;
  const _Float16* p = baseT + col * ldT + kb;
  v16h b;
#pragma unroll
  for (int q = 0; q < 16; ++q) b[q] = p[q];
  return b;
}

static __device__ __forceinline__ v8f wmma32(v16h a, v16h b, v8f c) {
  return __builtin_amdgcn_wmma_f32_16x16x32_f16(false, a, false, b, (short)0, c,
                                                false, false);
}

// ---- weight conversion: src row-major [srcK x srcN] f32 -> dstT [dstN x ldT] f16,
// zero-padded in both K (rows >= srcK) and N (cols >= srcN)
__global__ void k_convT(const float* __restrict__ src, _Float16* __restrict__ dstT,
                        int srcK, int srcN, int dstN, int ldT) {
  int i = blockIdx.x * blockDim.x + threadIdx.x;
  if (i >= dstN * ldT) return;
  int c = i / ldT, r = i - c * ldT;
  float v = (r < srcK && c < srcN) ? src[(size_t)r * srcN + c] : 0.0f;
  dstT[i] = (_Float16)v;
}

// ---- embedding gather: x row-major f32/f16 + feature-major f16 mirror
__global__ void k_embed(const int* __restrict__ species, const float* __restrict__ emb,
                        float* __restrict__ x32, _Float16* __restrict__ x16,
                        _Float16* __restrict__ x16T) {
  int i = blockIdx.x * blockDim.x + threadIdx.x;
  if (i >= kB * kN * kF) return;
  int f  = i & (kF - 1);
  int bn = i >> 7;
  int b  = bn >> 8, n = bn & (kN - 1);
  float v = emb[species[bn] * kF + f];
  x32[i] = v;
  x16[i] = (_Float16)v;
  x16T[((size_t)b * kF + f) * kN + n] = (_Float16)v;
}

// ---- conv einsum: m[b,i,k,f] = sum_j g[b,i,j,k]*x[b,j,f]
// one block per (b,i); g tile built in LDS once, shared by all 8 f-tiles
__global__ __launch_bounds__(256) void k_conv_einsum(
    const float* __restrict__ coords, const _Float16* __restrict__ x16T,
    const float* __restrict__ mu, const float* __restrict__ w,
    _Float16* __restrict__ mH) {
  int bi = blockIdx.x;
  int b  = bi / kN, i = bi & (kN - 1);
  int t  = threadIdx.x, wv = t >> 5, lane = t & 31;

  __shared__ float dL[kN];
  __shared__ float muL[kK], wL[kK];
  __shared__ __align__(16) _Float16 gL[kK][32];     // A tile [k][j], row-major
  __shared__ __align__(16) _Float16 xLT[kF][32];    // B tile K-major [f][j]
  __shared__ __align__(16) _Float16 mS[kK][kF];     // output staging

  if (t < kK) { muL[t] = mu[t]; wL[t] = w[t]; }
  {
    float cx = coords[bi * 3 + 0], cy = coords[bi * 3 + 1], cz = coords[bi * 3 + 2];
    int j = t;
    float dx = cx - coords[(b * kN + j) * 3 + 0];
    float dy = cy - coords[(b * kN + j) * 3 + 1];
    float dz = cz - coords[(b * kN + j) * 3 + 2];
    dL[j] = sqrtf(dx * dx + dy * dy + dz * dz + 1e-9f);
  }
  __syncthreads();

  int ft = wv;                       // f tile 0..7
  v8f acc0 = {}, acc1 = {};          // k rows 0-15 / 16-31

  for (int j0 = 0; j0 < kN; j0 += 32) {
#pragma unroll
    for (int q = 0; q < 4; ++q) {    // build g tile (32k x 32j)
      int idx = t * 4 + q;
      int k = idx >> 5, j = idx & 31;
      float z  = (dL[j0 + j] - muL[k]) / wL[k];
      float gv = __expf(-z * z);
      if (j0 + j == i) gv = 0.0f;    // mask self-interaction
      gL[k][j] = (_Float16)gv;
    }
    {                                // stage x tile K-major: [128f][32j]
      int f = t >> 1, c0 = (t & 1) * 16;
      const _Float16* src = x16T + ((size_t)b * kF + f) * kN + j0 + c0;
      __builtin_prefetch(src + 32, 0, 0);
#pragma unroll
      for (int q = 0; q < 16; ++q) xLT[f][c0 + q] = src[q];
    }
    __syncthreads();
    v16h bf = loadBT(&xLT[ft * 16][0], 32);
    v16h a0 = loadA(&gL[0][0], 32);
    v16h a1 = loadA(&gL[16][0], 32);
    acc0 = wmma32(a0, bf, acc0);
    acc1 = wmma32(a1, bf, acc1);
    __syncthreads();
  }

  // stage results in LDS, then contiguous b128 global stores
  {
    int col = ft * 16 + (lane & 15);
    int rb  = (lane & 16) ? 8 : 0;
#pragma unroll
    for (int r = 0; r < 8; ++r) {
      mS[rb + r][col]      = (_Float16)acc0[r];
      mS[16 + rb + r][col] = (_Float16)acc1[r];
    }
  }
  __syncthreads();
  {
    const _Float16* s = &mS[0][0] + t * 16;
    _Float16* d = mH + (size_t)bi * (kK * kF) + t * 16;
#pragma unroll
    for (int q = 0; q < 16; ++q) d[q] = s[q];
  }
}

// ---- h1 = silu(m @ Wc1 + bc1) : [8192 x 4096] @ [4096 x 256], Wc1 K-major
__global__ __launch_bounds__(256) void k_mlp1(
    const _Float16* __restrict__ mH, const _Float16* __restrict__ Wc1T,
    const float* __restrict__ bc1, _Float16* __restrict__ h1) {
  int row0 = blockIdx.x * 64, col0 = blockIdx.y * 32;
  int t = threadIdx.x, wv = t >> 5, lane = t & 31;
  int wr = wv >> 1, wc = wv & 1;
  __shared__ __align__(16) _Float16 AL[64][32];
  v8f acc = {};
  for (int kk = 0; kk < kK * kF; kk += 32) {
    {
      int r = t >> 2, c0 = (t & 3) * 8;
      const _Float16* src = mH + (size_t)(row0 + r) * (kK * kF) + kk + c0;
      __builtin_prefetch(src + 32, 0, 0);
#pragma unroll
      for (int q = 0; q < 8; ++q) AL[r][c0 + q] = src[q];
    }
    __syncthreads();
    v16h a  = loadA(&AL[wr * 16][0], 32);
    v16h bf = loadBT(Wc1T + (size_t)(col0 + wc * 16) * (kK * kF) + kk, kK * kF);
    acc = wmma32(a, bf, acc);
    __syncthreads();
  }
  // stage tile in LDS (reuse AL), then contiguous stores
  {
    int col = wc * 16 + (lane & 15);
    int rb  = (lane & 16) ? 8 : 0;
    float bias = bc1[col0 + col];
#pragma unroll
    for (int r = 0; r < 8; ++r)
      AL[wr * 16 + rb + r][col] = (_Float16)siluf(acc[r] + bias);
  }
  __syncthreads();
  {
    int r = t >> 2, c0 = (t & 3) * 8;
    _Float16* d = h1 + (size_t)(row0 + r) * kH + col0 + c0;
#pragma unroll
    for (int q = 0; q < 8; ++q) d[q] = AL[r][c0 + q];
  }
}

// ---- x = (1-u)x + u*scale*(h1 @ Wc2 + bc2), Wc2 K-major; updates x32/x16/x16T
__global__ __launch_bounds__(256) void k_mlp2_update(
    const _Float16* __restrict__ h1, const _Float16* __restrict__ Wc2T,
    const float* __restrict__ bc2, float* __restrict__ x32,
    _Float16* __restrict__ x16, _Float16* __restrict__ x16T, float scale) {
  int row0 = blockIdx.x * 64, col0 = blockIdx.y * 32;
  int t = threadIdx.x, wv = t >> 5, lane = t & 31;
  int wr = wv >> 1, wc = wv & 1;
  __shared__ __align__(16) _Float16 AL[64][32];
  v8f acc = {};
  for (int kk = 0; kk < kH; kk += 32) {
    {
      int r = t >> 2, c0 = (t & 3) * 8;
      const _Float16* src = h1 + (size_t)(row0 + r) * kH + kk + c0;
#pragma unroll
      for (int q = 0; q < 8; ++q) AL[r][c0 + q] = src[q];
    }
    __syncthreads();
    v16h a  = loadA(&AL[wr * 16][0], 32);
    v16h bf = loadBT(Wc2T + (size_t)(col0 + wc * 16) * kH + kk, kH);
    acc = wmma32(a, bf, acc);
    __syncthreads();
  }
  int col = col0 + wc * 16 + (lane & 15);
  int rb  = (lane & 16) ? 8 : 0;
  float bias = bc2[col];
#pragma unroll
  for (int r = 0; r < 8; ++r) {
    int row = row0 + wr * 16 + rb + r;
    size_t idx = (size_t)row * kF + col;
    float nv = (1.0f - kUPD) * x32[idx] + kUPD * scale * (acc[r] + bias);
    x32[idx] = nv;
    x16[idx] = (_Float16)nv;
    int b = row >> 8, n = row & (kN - 1);
    x16T[((size_t)b * kF + col) * kN + n] = (_Float16)nv;
  }
}

// ---- c_iso = silu(x@Wp1+bp1)@Wp2+bp2 (weights K-major)
__global__ __launch_bounds__(256) void k_pool(
    const _Float16* __restrict__ x16, const _Float16* __restrict__ Wp1T,
    const float* __restrict__ bp1, const _Float16* __restrict__ Wp2T,
    const float* __restrict__ bp2, float* __restrict__ outCiso) {
  int row0 = blockIdx.x * 16;
  int t = threadIdx.x, wv = t >> 5, lane = t & 31;
  __shared__ __align__(16) _Float16 XL[16][kF];
  __shared__ __align__(16) _Float16 HL[16][kHP];
  {
    int r = t >> 4, c0 = (t & 15) * 8;
    const _Float16* src = x16 + (size_t)(row0 + r) * kF + c0;
#pragma unroll
    for (int q = 0; q < 8; ++q) XL[r][c0 + q] = src[q];
  }
  __syncthreads();
  v8f acc = {};
  for (int kk = 0; kk < kF; kk += 32) {
    v16h a  = loadA(&XL[0][kk], kF);
    v16h bf = loadBT(Wp1T + (size_t)(wv * 16) * kF + kk, kF);
    acc = wmma32(a, bf, acc);
  }
  {
    int col = wv * 16 + (lane & 15);
    int rb  = (lane & 16) ? 8 : 0;
    float bias = bp1[col];
#pragma unroll
    for (int r = 0; r < 8; ++r) HL[rb + r][col] = (_Float16)siluf(acc[r] + bias);
  }
  __syncthreads();
  if (wv == 0) {
    v8f acc2 = {};
    for (int kk = 0; kk < kHP; kk += 32) {
      v16h a  = loadA(&HL[0][kk], kHP);
      v16h bf = loadBT(Wp2T + kk, kHP);
      acc2 = wmma32(a, bf, acc2);
    }
    int col = lane & 15;
    int rb  = (lane & 16) ? 8 : 0;
    float bias = bp2[col];
#pragma unroll
    for (int r = 0; r < 8; ++r)
      outCiso[(size_t)(row0 + rb + r) * 16 + col] = acc2[r] + bias;
  }
}

// ---- 16-NN top-k: one wave per atom
__global__ __launch_bounds__(256) void k_topk(
    const float* __restrict__ coords, int* __restrict__ nidx, float* __restrict__ nd) {
  int wv = threadIdx.x >> 5, lane = threadIdx.x & 31;
  int bi = blockIdx.x * 8 + wv;
  int b  = bi / kN, i = bi & (kN - 1);
  float cx = coords[bi * 3 + 0], cy = coords[bi * 3 + 1], cz = coords[bi * 3 + 2];
  float dv[8];
  int base = lane * 8;
#pragma unroll
  for (int q = 0; q < 8; ++q) {
    int j = base + q;
    float dx = cx - coords[(b * kN + j) * 3 + 0];
    float dy = cy - coords[(b * kN + j) * 3 + 1];
    float dz = cz - coords[(b * kN + j) * 3 + 2];
    float d  = sqrtf(dx * dx + dy * dy + dz * dz + 1e-9f);
    dv[q] = (j == i) ? kINF : d;
  }
  for (int sel = 0; sel < kKNB; ++sel) {
    float bd = kINF; int bj = 0x7FFFFFFF;
#pragma unroll
    for (int q = 0; q < 8; ++q)
      if (dv[q] < bd) { bd = dv[q]; bj = base + q; }
#pragma unroll
    for (int off = 16; off > 0; off >>= 1) {
      float od = __shfl_xor(bd, off, 32);
      int   oj = __shfl_xor(bj, off, 32);
      if (od < bd || (od == bd && oj < bj)) { bd = od; bj = oj; }
    }
#pragma unroll
    for (int q = 0; q < 8; ++q)
      if (bj == base + q) dv[q] = kINF;
    if (lane == 0) {
      nidx[(size_t)bi * kKNB + sel] = bj;
      nd[(size_t)bi * kKNB + sel]   = bd;
    }
  }
}

// ---- edge MLPs + attention (weights K-major)
__global__ __launch_bounds__(256) void k_edge(
    const _Float16* __restrict__ x16, const int* __restrict__ nidx,
    const float* __restrict__ nd,
    const _Float16* __restrict__ Wa1T, const float* __restrict__ ba1,
    const _Float16* __restrict__ Wa2T, const float* __restrict__ ba2,
    const _Float16* __restrict__ We1T, const float* __restrict__ be1,
    const _Float16* __restrict__ We2T, const float* __restrict__ be2,
    float* __restrict__ outAniso) {
  int bi = blockIdx.x;
  int b  = bi / kN;
  int t  = threadIdx.x, wv = t >> 5, lane = t & 31;

  __shared__ __align__(16) _Float16 PL[kKNB][kPairP];
  __shared__ __align__(16) _Float16 HAL[kKNB][kHA];
  __shared__ __align__(16) _Float16 HEL[kKNB][kHE];
  __shared__ float LG[kKNB][16];
  __shared__ float EV[kKNB][kM * kC];
  __shared__ float AT[kKNB][kM];
  __shared__ float DN[kKNB];
  __shared__ int   JJ[kKNB];

  if (t < kKNB) { JJ[t] = nidx[(size_t)bi * kKNB + t]; DN[t] = nd[(size_t)bi * kKNB + t]; }
  __syncthreads();

  for (int idx = t; idx < kKNB * kPairP; idx += 256) {
    int r = idx / kPairP, c = idx - r * kPairP;
    _Float16 v;
    if (c < kF)            v = x16[(size_t)bi * kF + c];
    else if (c < 2 * kF)   v = x16[(size_t)(b * kN + JJ[r]) * kF + (c - kF)];
    else if (c == 2 * kF)  v = (_Float16)DN[r];
    else                   v = (_Float16)0.0f;
    PL[r][c] = v;
  }
  __syncthreads();

  {  // he hidden: 8 tiles, one per wave
    v8f acc = {};
    for (int kk = 0; kk < kPairP; kk += 32) {
      v16h a  = loadA(&PL[0][kk], kPairP);
      v16h bf = loadBT(We1T + (size_t)(wv * 16) * kPairP + kk, kPairP);
      acc = wmma32(a, bf, acc);
    }
    int col = wv * 16 + (lane & 15);
    int rb  = (lane & 16) ? 8 : 0;
    float bias = be1[col];
#pragma unroll
    for (int r = 0; r < 8; ++r) HEL[rb + r][col] = (_Float16)siluf(acc[r] + bias);
  }
  if (wv < 4) {  // ha hidden: 4 tiles on waves 0-3
    v8f acc = {};
    for (int kk = 0; kk < kPairP; kk += 32) {
      v16h a  = loadA(&PL[0][kk], kPairP);
      v16h bf = loadBT(Wa1T + (size_t)(wv * 16) * kPairP + kk, kPairP);
      acc = wmma32(a, bf, acc);
    }
    int col = wv * 16 + (lane & 15);
    int rb  = (lane & 16) ? 8 : 0;
    float bias = ba1[col];
#pragma unroll
    for (int r = 0; r < 8; ++r) HAL[rb + r][col] = (_Float16)siluf(acc[r] + bias);
  }
  __syncthreads();

  if (wv == 0) {  // logits = ha @ Wa2 (cols padded to 16)
    v8f acc = {};
    for (int kk = 0; kk < kHA; kk += 32) {
      v16h a  = loadA(&HAL[0][kk], kHA);
      v16h bf = loadBT(Wa2T + kk, kHA);
      acc = wmma32(a, bf, acc);
    }
    int col = lane & 15;
    int rb  = (lane & 16) ? 8 : 0;
    float bias = (col < kM) ? ba2[col] : 0.0f;
#pragma unroll
    for (int r = 0; r < 8; ++r) LG[rb + r][col] = acc[r] + bias;
  }
  if (wv == 1 || wv == 2) {  // e = he @ We2 (32 cols = 2 tiles)
    int ct = wv - 1;
    v8f acc = {};
    for (int kk = 0; kk < kHE; kk += 32) {
      v16h a  = loadA(&HEL[0][kk], kHE);
      v16h bf = loadBT(We2T + (size_t)(ct * 16) * kHE + kk, kHE);
      acc = wmma32(a, bf, acc);
    }
    int col = ct * 16 + (lane & 15);
    int rb  = (lane & 16) ? 8 : 0;
    float bias = be2[col];
#pragma unroll
    for (int r = 0; r < 8; ++r) EV[rb + r][col] = acc[r] + bias;
  }
  __syncthreads();

  if (t < kM) {  // masked softmax over the 16 neighbors
    int m = t;
    float mx = -kINF;
    for (int nb = 0; nb < kKNB; ++nb) {
      float lg = (DN[nb] < kRC) ? LG[nb][m] : -1e9f;
      LG[nb][m] = lg;
      mx = fmaxf(mx, lg);
    }
    float s = 0.0f;
    for (int nb = 0; nb < kKNB; ++nb) {
      float e = __expf(LG[nb][m] - mx);
      AT[nb][m] = e;
      s += e;
    }
    float inv = 1.0f / s;
    for (int nb = 0; nb < kKNB; ++nb) AT[nb][m] *= inv;
  }
  __syncthreads();

  for (int idx = t; idx < kKNB * kM * kC; idx += 256) {
    int nb = idx >> 5;
    int mc = idx & 31;
    int m  = mc >> 3;
    outAniso[(size_t)bi * (kKNB * kM * kC) + idx] = AT[nb][m] * EV[nb][mc];
  }
}

extern "C" void kernel_launch(void* const* d_in, const int* in_sizes, int n_in,
                              void* d_out, int out_size, void* d_ws, size_t ws_size,
                              hipStream_t stream) {
  (void)in_sizes; (void)n_in; (void)out_size; (void)ws_size;
  const int*   species = (const int*)d_in[0];
  const float* coords  = (const float*)d_in[1];
  const float* emb     = (const float*)d_in[2];
  const float* mu      = (const float*)d_in[3];
  const float* w       = (const float*)d_in[4];
  const float* Wc1 = (const float*)d_in[5];  const float* bc1 = (const float*)d_in[6];
  const float* Wc2 = (const float*)d_in[7];  const float* bc2 = (const float*)d_in[8];
  const float* Wp1 = (const float*)d_in[9];  const float* bp1 = (const float*)d_in[10];
  const float* Wp2 = (const float*)d_in[11]; const float* bp2 = (const float*)d_in[12];
  const float* Wa1 = (const float*)d_in[13]; const float* ba1 = (const float*)d_in[14];
  const float* Wa2 = (const float*)d_in[15]; const float* ba2 = (const float*)d_in[16];
  const float* We1 = (const float*)d_in[17]; const float* be1 = (const float*)d_in[18];
  const float* We2 = (const float*)d_in[19]; const float* be2 = (const float*)d_in[20];

  char* ws = (char*)d_ws;
  size_t off = 0;
  auto take = [&](size_t bytes) -> char* {
    char* p = ws + off;
    off = (off + bytes + 255) & ~(size_t)255;
    return p;
  };
  float*    x32  = (float*)take((size_t)kB * kN * kF * 4);
  _Float16* x16  = (_Float16*)take((size_t)kB * kN * kF * 2);
  _Float16* x16T = (_Float16*)take((size_t)kB * kN * kF * 2);
  _Float16* h1   = (_Float16*)take((size_t)kB * kN * kH * 2);
  _Float16* Wc1T = (_Float16*)take((size_t)kH * (kK * kF) * 2);
  _Float16* Wc2T = (_Float16*)take((size_t)kF * kH * 2);
  _Float16* Wp1T = (_Float16*)take((size_t)kHP * kF * 2);
  _Float16* Wp2T = (_Float16*)take((size_t)16 * kHP * 2);
  _Float16* Wa1T = (_Float16*)take((size_t)kHA * kPairP * 2);
  _Float16* Wa2T = (_Float16*)take((size_t)16 * kHA * 2);
  _Float16* We1T = (_Float16*)take((size_t)kHE * kPairP * 2);
  _Float16* We2T = (_Float16*)take((size_t)(kM * kC) * kHE * 2);
  int*      nidx = (int*)take((size_t)kB * kN * kKNB * 4);
  float*    ndst = (float*)take((size_t)kB * kN * kKNB * 4);
  _Float16* mH   = (_Float16*)take((size_t)kB * kN * kK * kF * 2);

  auto cdiv = [](int a, int b) { return (a + b - 1) / b; };

  // weights -> f16 K-major (transposed), K/N zero-padded where needed
  k_convT<<<cdiv(kH * kK * kF, 256), 256, 0, stream>>>(Wc1, Wc1T, kK * kF, kH, kH, kK * kF);
  k_convT<<<cdiv(kF * kH, 256), 256, 0, stream>>>(Wc2, Wc2T, kH, kF, kF, kH);
  k_convT<<<cdiv(kHP * kF, 256), 256, 0, stream>>>(Wp1, Wp1T, kF, kHP, kHP, kF);
  k_convT<<<cdiv(16 * kHP, 256), 256, 0, stream>>>(Wp2, Wp2T, kHP, 16, 16, kHP);
  k_convT<<<cdiv(kHA * kPairP, 256), 256, 0, stream>>>(Wa1, Wa1T, kPair, kHA, kHA, kPairP);
  k_convT<<<cdiv(16 * kHA, 256), 256, 0, stream>>>(Wa2, Wa2T, kHA, kM, 16, kHA);
  k_convT<<<cdiv(kHE * kPairP, 256), 256, 0, stream>>>(We1, We1T, kPair, kHE, kHE, kPairP);
  k_convT<<<cdiv(kM * kC * kHE, 256), 256, 0, stream>>>(We2, We2T, kHE, kM * kC, kM * kC, kHE);

  k_embed<<<cdiv(kB * kN * kF, 256), 256, 0, stream>>>(species, emb, x32, x16, x16T);

  float scale = 1.0f;
  for (int it = 0; it < 3; ++it) {
    k_conv_einsum<<<kB * kN, 256, 0, stream>>>(coords, x16T, mu, w, mH);
    k_mlp1<<<dim3(kB * kN / 64, kH / 32), 256, 0, stream>>>(mH, Wc1T, bc1, h1);
    k_mlp2_update<<<dim3(kB * kN / 64, kF / 32), 256, 0, stream>>>(h1, Wc2T, bc2,
                                                                   x32, x16, x16T, scale);
    scale *= 0.5f;
  }

  k_pool<<<kB * kN / 16, 256, 0, stream>>>(x16, Wp1T, bp1, Wp2T, bp2, (float*)d_out);
  k_topk<<<kB * kN / 8, 256, 0, stream>>>(coords, nidx, ndst);
  k_edge<<<kB * kN, 256, 0, stream>>>(x16, nidx, ndst, Wa1T, ba1, Wa2T, ba2,
                                      We1T, be1, We2T, be2,
                                      (float*)d_out + (size_t)kB * kN * 16);
}
